// SAGE_43997644980791
// MI455X (gfx1250) — matrix operand order repaired and checked by reading, not compiled, not run on hardware
//
#include <hip/hip_runtime.h>
#include <cstdint>

// ---------------------------------------------------------------------------
// GraphSAGE event-camera network forward pass for MI455X (gfx1250, wave32).
// All dense matmuls go through v_wmma_f32_16x16x32_f16; segment ops via atomics.
// ---------------------------------------------------------------------------

typedef __attribute__((ext_vector_type(16))) _Float16     v16h;
typedef __attribute__((ext_vector_type(8)))  float        v8f;
typedef __attribute__((ext_vector_type(4)))  unsigned int u32x4;

#define NN      131072      // nodes
#define EE      2097152     // edges
#define NGRAPH  8
#define NCELL   225
#define CC      1800        // coarse nodes (8 * 225)
#define CP      1808        // CC rounded up to 16
#define CPK     1824        // CC rounded up to 32
#define ENC_NEGINF 0x007FFFFFu

static inline int cdiv_i(long long a, long long b) { return (int)((a + b - 1) / b); }

__device__ inline unsigned encf(float f) {
  unsigned u = __float_as_uint(f);
  return (u & 0x80000000u) ? ~u : (u | 0x80000000u);
}
__device__ inline float decf(unsigned u) {
  unsigned b = (u & 0x80000000u) ? (u & 0x7FFFFFFFu) : ~u;
  return __uint_as_float(b);
}
__device__ inline int clampi(int v, int lo, int hi) { return v < lo ? lo : (v > hi ? hi : v); }

// ------------------------------- elementwise kernels ------------------------

__global__ void k_fill_u32(unsigned* p, unsigned v, long long n) {
  long long i = (long long)blockIdx.x * blockDim.x + threadIdx.x;
  if (i < n) p[i] = v;
}

__global__ void k_count_deg(const int* __restrict__ dst, float* deg, int E) {
  int i = blockIdx.x * blockDim.x + threadIdx.x;
  if (i < E) atomicAdd(&deg[dst[i]], 1.0f);
}

__global__ void k_inv(float* d, int n) {
  int i = blockIdx.x * blockDim.x + threadIdx.x;
  if (i < n) d[i] = 1.0f / fmaxf(d[i], 1.0f);
}

// scatter-add neighbor features: agg[dst] += h[src]
__global__ void k_scatter(const float* __restrict__ hin, const int* __restrict__ src,
                          const int* __restrict__ dst, float* agg, int E, int cin) {
  int e = blockIdx.x * blockDim.x + threadIdx.x;
  if (e >= E) return;
  int s = src[e], d = dst[e];
  const float* hs = hin + (size_t)s * cin;
  float* ad = agg + (size_t)d * cin;
  for (int c = 0; c < cin; ++c) atomicAdd(&ad[c], hs[c]);
}

// mean (agg * invdeg) -> f16 with K padded to Kp (zeros)
__global__ void k_scale_cvt(const float* __restrict__ agg, const float* __restrict__ invdeg,
                            _Float16* h16, int N, int cin, int Kp) {
  long long i = (long long)blockIdx.x * blockDim.x + threadIdx.x;
  if (i >= (long long)N * Kp) return;
  int n = (int)(i / Kp), c = (int)(i % Kp);
  float v = (c < cin) ? agg[(size_t)n * cin + c] * invdeg[n] : 0.0f;
  h16[i] = (_Float16)v;
}

// weight f32 [nout x nin] -> f16 [noutPad x Kp], zero padded (B-operand layout W[n][k])
__global__ void k_cvt_w(const float* __restrict__ W, _Float16* wb,
                        int nout, int nin, int noutPad, int Kp) {
  long long i = (long long)blockIdx.x * blockDim.x + threadIdx.x;
  if (i >= (long long)noutPad * Kp) return;
  int r = (int)(i / Kp), c = (int)(i % Kp);
  float v = (r < nout && c < nin) ? W[(size_t)r * nin + c] : 0.0f;
  wb[i] = (_Float16)v;
}

__global__ void k_finalize_stats(float* stats, int ncol, float cnt) {
  int n = blockIdx.x * blockDim.x + threadIdx.x;
  if (n >= ncol) return;
  float mean = stats[n] / cnt;
  float var  = stats[ncol + n] / cnt - mean * mean;
  stats[n]        = mean;
  stats[ncol + n] = rsqrtf(var + 1e-5f);
}

__global__ void k_bn(float* y, const float* __restrict__ stats,
                     const float* __restrict__ g, const float* __restrict__ b,
                     long long total, int ncol) {
  long long i = (long long)blockIdx.x * blockDim.x + threadIdx.x;
  if (i >= total) return;
  int n = (int)(i % ncol);
  y[i] = (y[i] - stats[n]) * stats[ncol + n] * g[n] + b[n];
}

__global__ void k_add(float* a, const float* __restrict__ bsrc, long long n) {
  long long i = (long long)blockIdx.x * blockDim.x + threadIdx.x;
  if (i < n) a[i] += bsrc[i];
}

// ------------------------------- pooling kernels ----------------------------

__global__ void k_cluster(const float* __restrict__ pos, const int* __restrict__ batch,
                          int* clu, float* cnt, float* sx, float* sy, int N) {
  int i = blockIdx.x * blockDim.x + threadIdx.x;
  if (i >= N) return;
  float px = pos[(size_t)i * 3], py = pos[(size_t)i * 3 + 1];
  int cx = clampi((int)floorf(px * (1.0f / 16.0f)), 0, 14);
  int cy = clampi((int)floorf(py * (1.0f / 12.0f)), 0, 14);
  int c = batch[i] * NCELL + cy * 15 + cx;
  clu[i] = c;
  atomicAdd(&cnt[c], 1.0f);
  atomicAdd(&sx[c], px);
  atomicAdd(&sy[c], py);
}

__global__ void k_segmax(const float* __restrict__ h, const int* __restrict__ clu,
                         unsigned* hpU, long long total) {  // total = N*128
  long long i = (long long)blockIdx.x * blockDim.x + threadIdx.x;
  if (i >= total) return;
  int n = (int)(i >> 7), ch = (int)(i & 127);
  atomicMax(&hpU[(size_t)clu[n] * 128 + ch], encf(h[i]));
}

__global__ void k_decode_hp(const unsigned* __restrict__ hpU, const float* __restrict__ cnt,
                            float* hp, long long total) {   // total = CP*128
  long long i = (long long)blockIdx.x * blockDim.x + threadIdx.x;
  if (i >= total) return;
  int c = (int)(i >> 7);
  float v = 0.0f;
  if (c < CC && cnt[c] > 0.0f) v = decf(hpU[i]);
  hp[i] = v;
}

__global__ void k_build_A(const int* __restrict__ src, const int* __restrict__ dst,
                          const int* __restrict__ clu, _Float16* A16, int E) {
  int e = blockIdx.x * blockDim.x + threadIdx.x;
  if (e >= E) return;
  int cs = clu[src[e]], cd = clu[dst[e]];
  if (cs != cd) A16[(size_t)cd * CPK + cs] = (_Float16)1.0f;   // coalesced 0/1, races benign
}

__global__ void k_degA(const _Float16* __restrict__ A16, float* invA, int Cp) {
  int m = blockIdx.x * blockDim.x + threadIdx.x;
  if (m >= Cp) return;
  float s = 0.0f;
  if (m < CC) {
    const _Float16* row = A16 + (size_t)m * CPK;
    for (int c = 0; c < CC; ++c) s += (float)row[c];
  }
  invA[m] = (m < CC) ? 1.0f / fmaxf(s, 1.0f) : 0.0f;
}

// hT[ch][c] = h[c][ch] (f32 -> f16, pad c>=CC with zeros) : B-operand for A@h
__global__ void k_tcvt(const float* __restrict__ h, _Float16* hT, long long total) { // 128*CPK
  long long i = (long long)blockIdx.x * blockDim.x + threadIdx.x;
  if (i >= total) return;
  int ch = (int)(i / CPK), c = (int)(i % CPK);
  float v = (c < CC) ? h[(size_t)c * 128 + ch] : 0.0f;
  hT[i] = (_Float16)v;
}

__global__ void k_pool7(const float* __restrict__ h, const float* __restrict__ sx,
                        const float* __restrict__ sy, const float* __restrict__ cnt,
                        unsigned* xfU, float* cnt7, long long total) {   // CC*128
  long long i = (long long)blockIdx.x * blockDim.x + threadIdx.x;
  if (i >= total) return;
  int c = (int)(i >> 7), ch = (int)(i & 127);
  float ic = 1.0f / fmaxf(cnt[c], 1.0f);
  float px = sx[c] * ic, py = sy[c] * ic;
  int c7x = clampi((int)floorf(px * (1.0f / 60.0f)), 0, 3);
  int c7y = clampi((int)floorf(py * (1.0f / 45.0f)), 0, 3);
  int seg = (c / NCELL) * 16 + c7y * 4 + c7x;
  atomicMax(&xfU[(size_t)seg * 128 + ch], encf(h[i]));
  if (ch == 0) atomicAdd(&cnt7[seg], 1.0f);
}

__global__ void k_xf(const unsigned* __restrict__ xfU, const float* __restrict__ cnt7,
                     _Float16* xf16, int total) {   // 128*128
  int i = blockIdx.x * blockDim.x + threadIdx.x;
  if (i >= total) return;
  int seg = i >> 7, ch = i & 127;
  float v = (cnt7[seg] > 0.0f) ? decf(xfU[i]) : 0.0f;
  int b = seg >> 4, cell = seg & 15;
  xf16[(size_t)b * 2048 + cell * 128 + ch] = (_Float16)v;
}

// ------------------------------- WMMA GEMM ----------------------------------
// D[M x ncol] = A[M x K] * B[K x ncol], B given as Bw[n][k] (= W row-major).
// One wave per 16x16 output tile; v_wmma_f32_16x16x32_f16, f32 accumulate.
// EPI 0: plain f32 store (guarded m<Mreal, n<Nreal)
// EPI 1: ELU + f32 store + per-channel sum/sumsq atomics into stats[2*Nreal]
// EPI 2: f16 store with per-row scale (rowScale[m]) -> feeds next WMMA stage
template <int EPI>
__global__ void k_gemm(const _Float16* __restrict__ A, const _Float16* __restrict__ Bw,
                       int ldA, int ldB, int K, int tilesN, int totalTiles,
                       float* outF, _Float16* outH, int ldOut, int Mreal, int Nreal,
                       const float* __restrict__ rowScale, float* stats) {
  int wid = (int)(((long long)blockIdx.x * blockDim.x + threadIdx.x) >> 5);
  if (wid >= totalTiles) return;                 // wave-uniform: EXEC stays all ones
  int lane = threadIdx.x & 31;
  int hi   = lane >> 4;                          // lane half: 0 -> lanes 0-15, 1 -> 16-31
  int r    = lane & 15;
  int mTile = wid / tilesN, nTile = wid - mTile * tilesN;
  int m0 = mTile << 4, n0 = nTile << 4;

  // A frag: row m0+r; hi half owns K runs [kb,kb+8) and [16+kb,16+kb+8), kb = 8*hi
  const _Float16* arow = A  + (size_t)(m0 + r) * ldA + (hi << 3);
  // B frag: column n = n0+r (== row n of Bw); hi half owns K runs [kb,kb+8) and [16+kb,16+kb+8), kb = 8*hi
  const _Float16* brow = Bw + (size_t)(n0 + r) * ldB + (hi << 3);

  union CU { v8f v; float f[8]; } acc;
#pragma unroll
  for (int i = 0; i < 8; ++i) acc.f[i] = 0.0f;

  for (int k0 = 0; k0 < K; k0 += 32) {
    union AF { v16h v; u32x4 q[2]; } af, bf;
    af.q[0] = *reinterpret_cast<const u32x4*>(arow + k0);
    af.q[1] = *reinterpret_cast<const u32x4*>(arow + k0 + 16);
    const u32x4* pb = reinterpret_cast<const u32x4*>(brow + k0);
    bf.q[0] = pb[0];
    bf.q[1] = pb[2];
    acc.v = __builtin_amdgcn_wmma_f32_16x16x32_f16(false, af.v, false, bf.v,
                                                   (short)0, acc.v, false, false);
  }

  int n = n0 + r;
  if (EPI == 0) {
#pragma unroll
    for (int e = 0; e < 8; ++e) {
      int m = m0 + e + (hi << 3);
      if (m < Mreal && n < Nreal) outF[(size_t)m * ldOut + n] = acc.f[e];
    }
  } else if (EPI == 1) {
    float s1 = 0.0f, s2 = 0.0f;
#pragma unroll
    for (int e = 0; e < 8; ++e) {
      int m = m0 + e + (hi << 3);
      if (m < Mreal) {
        float x = acc.f[e];
        float y = x > 0.0f ? x : __expf(x) - 1.0f;     // ELU
        outF[(size_t)m * ldOut + n] = y;
        s1 += y;
        s2 += y * y;
      }
    }
    atomicAdd(&stats[n], s1);
    atomicAdd(&stats[Nreal + n], s2);
  } else {  // EPI == 2
#pragma unroll
    for (int e = 0; e < 8; ++e) {
      int m = m0 + e + (hi << 3);
      outH[(size_t)m * ldOut + n] = (_Float16)(acc.f[e] * rowScale[m]);
    }
  }
}

// ------------------------------- driver -------------------------------------

extern "C" void kernel_launch(void* const* d_in, const int* in_sizes, int n_in,
                              void* d_out, int out_size, void* d_ws, size_t ws_size,
                              hipStream_t stream) {
  (void)in_sizes; (void)n_in; (void)out_size; (void)ws_size;
  const float* x     = (const float*)d_in[0];
  const float* pos   = (const float*)d_in[1];
  const int*   ei    = (const int*)d_in[2];
  const int*   batch = (const int*)d_in[3];
  const int* src = ei;
  const int* dst = ei + EE;
  const float *W[7], *Gm[7], *Bt[7];
  for (int i = 0; i < 7; ++i) {
    W[i]  = (const float*)d_in[4 + 3 * i];
    Gm[i] = (const float*)d_in[5 + 3 * i];
    Bt[i] = (const float*)d_in[6 + 3 * i];
  }
  const float* Wfc = (const float*)d_in[25];

  // bump allocator over workspace (256B aligned)
  char* wp = (char*)d_ws;
  auto alloc = [&](size_t bytes) -> void* {
    void* r = (void*)wp;
    wp += (bytes + 255) & ~(size_t)255;
    return r;
  };
  float*    invdeg = (float*)alloc((size_t)NN * 4);
  int*      clu    = (int*)alloc((size_t)NN * 4);
  float*    bufA   = (float*)alloc((size_t)NN * 64 * 4);
  float*    bufB   = (float*)alloc((size_t)NN * 64 * 4);
  float*    bufC   = (float*)alloc((size_t)NN * 128 * 4);
  float*    AG     = (float*)alloc((size_t)NN * 64 * 4);
  _Float16* H16    = (_Float16*)alloc((size_t)NN * 64 * 2);
  _Float16* WB     = (_Float16*)alloc((size_t)112 * 2048 * 2);
  float*    stats  = (float*)alloc(2 * 128 * 4);
  float*    cnt    = (float*)alloc(CC * 4);
  float*    sx     = (float*)alloc(CC * 4);
  float*    sy     = (float*)alloc(CC * 4);
  unsigned* hpU    = (unsigned*)alloc((size_t)CC * 128 * 4);
  float*    hp     = (float*)alloc((size_t)CP * 128 * 4);
  _Float16* A16    = (_Float16*)alloc((size_t)CP * CPK * 2);
  float*    invA   = (float*)alloc(CP * 4);
  _Float16* hT16   = (_Float16*)alloc((size_t)128 * CPK * 2);
  _Float16* s16    = (_Float16*)alloc((size_t)CP * 128 * 2);
  float*    h2     = (float*)alloc((size_t)CP * 128 * 4);
  float*    h2b    = (float*)alloc((size_t)CP * 128 * 4);
  float*    cnt7   = (float*)alloc(128 * 4);
  unsigned* xfU    = (unsigned*)alloc(128 * 128 * 4);
  _Float16* xf16   = (_Float16*)alloc((size_t)16 * 2048 * 2);

  const int TB = 256;

  // in-degree and 1/max(deg,1)
  hipMemsetAsync(invdeg, 0, (size_t)NN * 4, stream);
  k_count_deg<<<cdiv_i(EE, TB), TB, 0, stream>>>(dst, invdeg, EE);
  k_inv<<<cdiv_i(NN, TB), TB, 0, stream>>>(invdeg, NN);

  // one fine SAGE layer: mean-agg -> f16 -> WMMA linear (+ELU+stats) -> BN
  auto fine_layer = [&](const float* hin, int cin, int cout, float* hout, int li) {
    int Kp = (cin + 31) & ~31;
    hipMemsetAsync(AG, 0, (size_t)NN * cin * 4, stream);
    k_scatter<<<cdiv_i(EE, TB), TB, 0, stream>>>(hin, src, dst, AG, EE, cin);
    k_scale_cvt<<<cdiv_i((long long)NN * Kp, TB), TB, 0, stream>>>(AG, invdeg, H16, NN, cin, Kp);
    k_cvt_w<<<cdiv_i((long long)cout * Kp, TB), TB, 0, stream>>>(W[li], WB, cout, cin, cout, Kp);
    hipMemsetAsync(stats, 0, (size_t)2 * cout * 4, stream);
    int tiles = (NN / 16) * (cout / 16);
    k_gemm<1><<<cdiv_i((long long)tiles * 32, TB), TB, 0, stream>>>(
        H16, WB, Kp, Kp, Kp, cout / 16, tiles, hout, nullptr, cout, NN, cout, nullptr, stats);
    k_finalize_stats<<<1, 128, 0, stream>>>(stats, cout, (float)NN);
    k_bn<<<cdiv_i((long long)NN * cout, TB), TB, 0, stream>>>(
        hout, stats, Gm[li], Bt[li], (long long)NN * cout, cout);
  };

  fine_layer(x,    1,  16,  bufA, 0);
  fine_layer(bufA, 16, 64,  bufB, 1);   // bufB = skip
  fine_layer(bufB, 64, 64,  bufC, 2);
  fine_layer(bufC, 64, 64,  bufA, 3);
  k_add<<<cdiv_i((long long)NN * 64, TB), TB, 0, stream>>>(bufA, bufB, (long long)NN * 64);
  fine_layer(bufA, 64, 128, bufC, 4);   // h5 in bufC

  // pool5: voxel clusters, counts, pos sums, segment-max features
  hipMemsetAsync(cnt, 0, CC * 4, stream);
  hipMemsetAsync(sx,  0, CC * 4, stream);
  hipMemsetAsync(sy,  0, CC * 4, stream);
  k_cluster<<<cdiv_i(NN, TB), TB, 0, stream>>>(pos, batch, clu, cnt, sx, sy, NN);
  k_fill_u32<<<cdiv_i((long long)CC * 128, TB), TB, 0, stream>>>(hpU, ENC_NEGINF, (long long)CC * 128);
  k_segmax<<<cdiv_i((long long)NN * 128, TB), TB, 0, stream>>>(bufC, clu, hpU, (long long)NN * 128);
  k_decode_hp<<<cdiv_i((long long)CP * 128, TB), TB, 0, stream>>>(hpU, cnt, hp, (long long)CP * 128);

  // coarse adjacency (coalesced 0/1, self-loops removed) + inverse row degree
  hipMemsetAsync(A16, 0, (size_t)CP * CPK * 2, stream);
  k_build_A<<<cdiv_i(EE, TB), TB, 0, stream>>>(src, dst, clu, A16, EE);
  k_degA<<<cdiv_i(CP, TB), TB, 0, stream>>>(A16, invA, CP);

  // coarse SAGE layer: (A@h)*invDeg via WMMA (EPI=2) then WMMA linear (EPI=1) + BN
  auto coarse_layer = [&](const float* hin, float* hout, int li) {
    k_tcvt<<<cdiv_i((long long)128 * CPK, TB), TB, 0, stream>>>(hin, hT16, (long long)128 * CPK);
    int tiles = (CP / 16) * 8;
    k_gemm<2><<<cdiv_i((long long)tiles * 32, TB), TB, 0, stream>>>(
        A16, hT16, CPK, CPK, CPK, 8, tiles, nullptr, s16, 128, CP, 128, invA, nullptr);
    k_cvt_w<<<cdiv_i((long long)128 * 128, TB), TB, 0, stream>>>(W[li], WB, 128, 128, 128, 128);
    hipMemsetAsync(stats, 0, 2 * 128 * 4, stream);
    k_gemm<1><<<cdiv_i((long long)tiles * 32, TB), TB, 0, stream>>>(
        s16, WB, 128, 128, 128, 8, tiles, hout, nullptr, 128, CC, 128, nullptr, stats);
    k_finalize_stats<<<1, 128, 0, stream>>>(stats, 128, (float)CC);
    k_bn<<<cdiv_i((long long)CC * 128, TB), TB, 0, stream>>>(
        hout, stats, Gm[li], Bt[li], (long long)CC * 128, 128);
  };
  coarse_layer(hp, h2, 5);
  coarse_layer(h2, h2b, 6);
  k_add<<<cdiv_i((long long)CC * 128, TB), TB, 0, stream>>>(h2b, hp, (long long)CC * 128);

  // pool7 (4x4 per graph) + final FC through WMMA into d_out
  hipMemsetAsync(cnt7, 0, 128 * 4, stream);
  k_fill_u32<<<cdiv_i(128 * 128, TB), TB, 0, stream>>>(xfU, ENC_NEGINF, 128 * 128);
  hipMemsetAsync(xf16, 0, (size_t)16 * 2048 * 2, stream);
  k_pool7<<<cdiv_i((long long)CC * 128, TB), TB, 0, stream>>>(
      h2b, sx, sy, cnt, xfU, cnt7, (long long)CC * 128);
  k_xf<<<cdiv_i(128 * 128, TB), TB, 0, stream>>>(xfU, cnt7, xf16, 128 * 128);
  k_cvt_w<<<cdiv_i((long long)112 * 2048, TB), TB, 0, stream>>>(Wfc, WB, 101, 2048, 112, 2048);
  k_gemm<0><<<1, TB, 0, stream>>>(xf16, WB, 2048, 2048, 2048, 7, 7,
                                  (float*)d_out, nullptr, 101, NGRAPH, 101, nullptr, nullptr);
}
